// TrajPreLocalAttnLong_4209067950205
// MI455X (gfx1250) — matrix operand order repaired and checked by reading, not compiled, not run on hardware
//
#include <hip/hip_runtime.h>
#include <hip/hip_bf16.h>
#include <math.h>

// ---------------------------------------------------------------------------
// TrajPreLocalAttnLong for MI455X (gfx1250, wave32, WMMA)
//   SEQ=4096, TGT=1024, SENC=3072, IN=544, HID=1024, 3H=3072, NF=50000
// ---------------------------------------------------------------------------

#define SEQ   4096
#define TGT   1024
#define SENC  3072
#define INF_  544
#define HID   1024
#define H3    3072
#define NF    50000
#define D2H   2048   // 2*HID

#define GRU_NB     32      // blocks per GRU (each owns 32 hidden units)
#define GRU_THREADS 384    // 96 rows x 4 k-quarters
// LDS: 96*512 uint (bf16x2 weights) + 1024 f32 h + 384 f32 psum + 96 f32 bias
#define GRU_LDS_BYTES (96 * 512 * 4 + 1024 * 4 + 384 * 4 + 96 * 4)

typedef __bf16 bf16;
typedef bf16  v16bf __attribute__((ext_vector_type(16)));
typedef float v8f   __attribute__((ext_vector_type(8)));
typedef int   v4i   __attribute__((ext_vector_type(4)));

#if __has_builtin(__builtin_amdgcn_global_load_async_to_lds_b128) && \
    __has_builtin(__builtin_amdgcn_s_wait_asynccnt)
#define USE_ASYNC_LDS 1
#else
#define USE_ASYNC_LDS 0
#endif

union Frag16 {
    v16bf v;
    uint4 q[2];
    bf16  e[16];
};

__device__ inline unsigned short bf16_bits(float f) {
    union { bf16 b; unsigned short s; } u;
    u.b = (bf16)f;
    return u.s;
}

// ---- WMMA fragment loaders (contiguous k = 16*(L>>4) + e, same order for A and B) -----

// A-matrix 16x32 bf16: lane L (m = base+(L&15)); e[0..15] = K kc+hi*16..+15
// contiguous, hi = L>>4.
__device__ inline v16bf load_frag_a(const bf16* __restrict__ A, int lda,
                                    int mbase, int kc, int lane) {
    Frag16 f;
    const bf16* p = A + (size_t)(mbase + (lane & 15)) * lda + kc + (lane >> 4) * 16;
    f.q[0] = *(const uint4*)(p);
    f.q[1] = *(const uint4*)(p + 8);
    return f.v;
}

// B-matrix 32x16 bf16 (row-major [N,K]): lane L holds column n = base+(L&15),
// K = kc+(L>>4)*16 .. +15 contiguous.
__device__ inline v16bf load_frag_b(const bf16* __restrict__ B, int ldb,
                                    int nbase, int kc, int lane) {
    Frag16 f;
    const bf16* p = B + (size_t)(nbase + (lane & 15)) * ldb + kc + (lane >> 4) * 16;
    f.q[0] = *(const uint4*)(p);
    f.q[1] = *(const uint4*)(p + 8);
    return f.v;
}

// Same as load_frag_b but source is f32 (convert on the fly), with N bound.
__device__ inline v16bf load_frag_b_f32(const float* __restrict__ B, int ldb,
                                        int nbase, int kc, int lane, int N) {
    Frag16 f;
    int row = nbase + (lane & 15);
    int ks  = kc + (lane >> 4) * 16;
    if (row < N) {
        const float4* p = (const float4*)(B + (size_t)row * ldb + ks);
#pragma unroll
        for (int c = 0; c < 4; ++c) {
            float4 t = p[c];
            f.e[c * 4 + 0] = (bf16)t.x;
            f.e[c * 4 + 1] = (bf16)t.y;
            f.e[c * 4 + 2] = (bf16)t.z;
            f.e[c * 4 + 3] = (bf16)t.w;
        }
    } else {
        f.q[0] = make_uint4(0u, 0u, 0u, 0u);
        f.q[1] = make_uint4(0u, 0u, 0u, 0u);
    }
    return f.v;
}

// ---------------------------------------------------------------------------
// Generic C = A * B^T (+bias) WMMA GEMM, software-pipelined.
//   Block tile 256(M) x 128(N); 8 waves as 2(M) x 4(N); wave tile 128x32
//   = 16 v8f accumulators (128 VGPRs).  M % 256 == 0; N bounds-checked;
//   K % 32 == 0.
// ---------------------------------------------------------------------------
template <bool B_F32, bool BIAS>
__global__ __launch_bounds__(256) void gemm_abt_kernel(
    const bf16* __restrict__ A, const void* __restrict__ Bp,
    const float* __restrict__ bias, float* __restrict__ C,
    int M, int N, int K) {
    const int lane  = threadIdx.x & 31;
    const int wave  = threadIdx.x >> 5;
    const int wm    = wave >> 2;                       // 0..1
    const int wn    = wave & 3;                        // 0..3
    const int mbase = blockIdx.y * 256 + wm * 128;
    const int nbase = blockIdx.x * 128 + wn * 32;

    const bf16*  Bb = (const bf16*)Bp;
    const float* Bf = (const float*)Bp;

    v8f acc[8][2];
#pragma unroll
    for (int ms = 0; ms < 8; ++ms)
#pragma unroll
        for (int j = 0; j < 2; ++j)
#pragma unroll
            for (int i = 0; i < 8; ++i) acc[ms][j][i] = 0.0f;

    // Prologue: B fragments for k-chunk 0.
    v16bf bcur0, bcur1;
    if constexpr (B_F32) {
        bcur0 = load_frag_b_f32(Bf, K, nbase, 0, lane, N);
        bcur1 = load_frag_b_f32(Bf, K, nbase + 16, 0, lane, N);
    } else {
        bcur0 = load_frag_b(Bb, K, nbase, 0, lane);
        bcur1 = load_frag_b(Bb, K, nbase + 16, 0, lane);
    }

    for (int kc = 0; kc < K; kc += 32) {
        const bool more = (kc + 32) < K;
        // Prefetch B two chunks ahead (gfx1250 global_prefetch_b8).
        if (kc + 64 < K) {
            if constexpr (B_F32) {
                int row = nbase + (lane & 15);
                if (row < N)
                    __builtin_prefetch((const void*)(Bf + (size_t)row * K + kc + 64), 0, 3);
            } else {
                __builtin_prefetch(
                    (const void*)(Bb + (size_t)(nbase + (lane & 15)) * K + kc + 64), 0, 3);
            }
        }
        // Fetch next k-chunk's B fragments early (in flight during WMMAs).
        v16bf bnext0, bnext1;
        if (more) {
            if constexpr (B_F32) {
                bnext0 = load_frag_b_f32(Bf, K, nbase, kc + 32, lane, N);
                bnext1 = load_frag_b_f32(Bf, K, nbase + 16, kc + 32, lane, N);
            } else {
                bnext0 = load_frag_b(Bb, K, nbase, kc + 32, lane);
                bnext1 = load_frag_b(Bb, K, nbase + 16, kc + 32, lane);
            }
        }

        // m-subtile loop with one-A-ahead pipelining.
        v16bf anext = load_frag_a(A, K, mbase, kc, lane);
#pragma unroll
        for (int ms = 0; ms < 8; ++ms) {
            v16bf acur = anext;
            if (ms < 7) anext = load_frag_a(A, K, mbase + (ms + 1) * 16, kc, lane);
            acc[ms][0] = __builtin_amdgcn_wmma_f32_16x16x32_bf16(
                false, acur, false, bcur0, (short)0, acc[ms][0], false, false);
            acc[ms][1] = __builtin_amdgcn_wmma_f32_16x16x32_bf16(
                false, acur, false, bcur1, (short)0, acc[ms][1], false, false);
        }
        if (more) { bcur0 = bnext0; bcur1 = bnext1; }
    }

    // C/D layout: element i -> (m = sub + (lane>>4)*8 + i, n = sub + (lane&15))
#pragma unroll
    for (int ms = 0; ms < 8; ++ms)
#pragma unroll
        for (int j = 0; j < 2; ++j) {
            int n = nbase + j * 16 + (lane & 15);
            if (n < N) {
                float bv = BIAS ? bias[n] : 0.0f;
#pragma unroll
                for (int i = 0; i < 8; ++i) {
                    int m = mbase + ms * 16 + (lane >> 4) * 8 + i;
                    C[(size_t)m * N + n] = acc[ms][j][i] + bv;
                }
            }
        }
}

// ---------------------------------------------------------------------------
// Elementwise f32 -> bf16 convert.
// ---------------------------------------------------------------------------
__global__ __launch_bounds__(256) void cvt_f32_bf16_kernel(
    const float* __restrict__ in, bf16* __restrict__ out, int n) {
    int i = blockIdx.x * 256 + threadIdx.x;
    if (i < n) out[i] = (bf16)in[i];
}

// ---------------------------------------------------------------------------
// Embedding gather: x[s] = concat(emb_loc[loc[s]], emb_tim[tim[s]])  (bf16)
// ---------------------------------------------------------------------------
__global__ __launch_bounds__(256) void embed_kernel(
    const int* __restrict__ loc, const int* __restrict__ tim,
    const float* __restrict__ emb_loc, const float* __restrict__ emb_tim,
    bf16* __restrict__ x) {
    int s = blockIdx.x;
    int li = loc[s];
    int ti = tim[s];
    const float* pl = emb_loc + (size_t)li * 512;
    const float* pt = emb_tim + (size_t)ti * 32;
    bf16* xr = x + (size_t)s * INF_;
    for (int j = threadIdx.x; j < INF_; j += 256)
        xr[j] = (bf16)((j < 512) ? pl[j] : pt[j - 512]);
}

// ---------------------------------------------------------------------------
// Zero the GRU sync area (counters + ping-pong h buffers) so every graph
// replay starts from identical state.
// ---------------------------------------------------------------------------
__global__ __launch_bounds__(256) void init_sync_kernel(int* __restrict__ cnt,
                                                        float* __restrict__ hbuf) {
    for (int j = threadIdx.x; j < 4 * HID; j += 256) hbuf[j] = 0.0f;
    if (threadIdx.x < 2) cnt[threadIdx.x] = 0;
}

// ---------------------------------------------------------------------------
// Parallel-persistent GRU scan.
//   64 blocks: 0..31 encoder, 32..63 decoder.  Block b owns 32 hidden units
//   (96 Whh rows), converted to bf16 ONCE into 192 KB of CDNA5 LDS.
//   Per step: h (4 KB) is broadcast-staged into LDS with DEV-scope
//   async-to-LDS loads, 96x4 threads compute 256-long dot products from LDS,
//   32 threads finish the gates, write the next ping-pong h buffer with
//   device-scope release stores, then all 32 blocks meet at an atomic-counter
//   barrier.  Sequential cost ~ L2 round trip per step instead of 12.6 MB.
// ---------------------------------------------------------------------------
__global__ __launch_bounds__(GRU_THREADS) void gru_scan_kernel(
    const float* __restrict__ gi_enc, const float* __restrict__ Whh_enc,
    const float* __restrict__ bhh_enc,
    const float* __restrict__ gi_dec, const float* __restrict__ Whh_dec,
    const float* __restrict__ bhh_dec,
    float* __restrict__ hist_enc, bf16* __restrict__ hist_enc_bf,
    float* __restrict__ hist_dec, bf16* __restrict__ hist_dec_bf,
    int* __restrict__ sync_cnt, float* __restrict__ hbuf) {
    extern __shared__ char smem_raw[];
    unsigned int* wlds   = (unsigned int*)smem_raw;                 // 96*512 u32
    float*        h_lds  = (float*)(smem_raw + 96 * 512 * 4);       // 1024 f32
    float*        psum   = h_lds + HID;                             // 384 f32
    float*        bias_l = psum + GRU_THREADS;                      // 96 f32

    const int tid = threadIdx.x;
    const int g   = blockIdx.x >> 5;        // 0 = enc, 1 = dec
    const int blk = blockIdx.x & 31;
    const int U0  = blk * 32;

    const float* gi   = g ? gi_dec : gi_enc;
    const float* Whh  = g ? Whh_dec : Whh_enc;
    const float* bhh  = g ? bhh_dec : bhh_enc;
    float*       hist = g ? hist_dec : hist_enc;
    bf16*        hb   = g ? hist_dec_bf : hist_enc_bf;
    const int    steps = g ? TGT : SENC;
    int*         cnt  = sync_cnt + g;
    float*       hA   = hbuf + g * 2 * HID;
    float*       hB   = hA + HID;

    // One-time: convert this block's Whh slice to bf16x2 in LDS.
    for (int idx = tid; idx < 96 * 512; idx += GRU_THREADS) {
        int r = idx >> 9;           // 0..95 : gate*32 + unit
        int j = idx & 511;
        const float* src =
            Whh + ((size_t)((r >> 5) * HID + U0 + (r & 31))) * HID + j * 2;
        wlds[idx] = (unsigned int)bf16_bits(src[0]) |
                    ((unsigned int)bf16_bits(src[1]) << 16);
    }
    if (tid < 96) bias_l[tid] = bhh[(tid >> 5) * HID + U0 + (tid & 31)];
    __syncthreads();

    const int r = tid >> 2;   // 0..95
    const int q = tid & 3;    // 0..3
    const unsigned int* wrow = wlds + (size_t)r * 512 + q * 128;

    for (int s = 0; s < steps; ++s) {
        const float* cur = (s & 1) ? hB : hA;
        float*       nxt = (s & 1) ? hA : hB;

        // Stage h (4 KB) into LDS, reading at DEV scope (fresh from L2).
#if USE_ASYNC_LDS
        if (tid < 256) {
            float* curp = const_cast<float*>(cur) + tid * 4;
            __builtin_amdgcn_global_load_async_to_lds_b128(
                (__attribute__((address_space(1))) v4i*)curp,
                (__attribute__((address_space(3))) v4i*)(h_lds + tid * 4),
                0, /*cpol: SCOPE_DEV*/ 16);
            __builtin_amdgcn_s_wait_asynccnt(0);
        }
#else
        for (int j = tid; j < HID; j += GRU_THREADS)
            h_lds[j] = __hip_atomic_load(cur + j, __ATOMIC_RELAXED,
                                         __HIP_MEMORY_SCOPE_AGENT);
#endif
        __syncthreads();

        // 96 rows x 4 quarters of 256-long dot products out of LDS.
        float acc = 0.0f;
        const float2* hp = (const float2*)h_lds + q * 128;
#pragma unroll 8
        for (int i = 0; i < 128; ++i) {
            unsigned int w = wrow[i];
            float2 hv = hp[i];
            acc = fmaf(__uint_as_float(w << 16), hv.x, acc);
            acc = fmaf(__uint_as_float(w & 0xffff0000u), hv.y, acc);
        }
        psum[tid] = acc;   // tid == r*4+q
        __syncthreads();

        // Finish gates for this block's 32 units.
        if (tid < 32) {
            int G = U0 + tid;
            float gr = bias_l[tid] +
                       psum[(tid) * 4] + psum[(tid) * 4 + 1] +
                       psum[(tid) * 4 + 2] + psum[(tid) * 4 + 3];
            int rz = 32 + tid;
            float gz = bias_l[rz] +
                       psum[rz * 4] + psum[rz * 4 + 1] +
                       psum[rz * 4 + 2] + psum[rz * 4 + 3];
            int rn = 64 + tid;
            float gn = bias_l[rn] +
                       psum[rn * 4] + psum[rn * 4 + 1] +
                       psum[rn * 4 + 2] + psum[rn * 4 + 3];

            const float* git = gi + (size_t)s * H3;
            float ir = git[G], iz = git[HID + G], inn = git[2 * HID + G];
            float rv = 1.0f / (1.0f + __expf(-(ir + gr)));
            float zv = 1.0f / (1.0f + __expf(-(iz + gz)));
            float nv = tanhf(inn + rv * gn);
            float hn = (1.0f - zv) * nv + zv * h_lds[G];

            hist[(size_t)s * HID + G] = hn;
            hb[(size_t)s * HID + G]   = (bf16)hn;
            __hip_atomic_store(nxt + G, hn, __ATOMIC_RELEASE,
                               __HIP_MEMORY_SCOPE_AGENT);
        }

        // 32-block barrier: monotonically increasing arrival counter.
        __syncthreads();
        if (tid == 0) {
            __threadfence();
            __hip_atomic_fetch_add(cnt, 1, __ATOMIC_ACQ_REL,
                                   __HIP_MEMORY_SCOPE_AGENT);
            const int target = (s + 1) * GRU_NB;
            while (__hip_atomic_load(cnt, __ATOMIC_ACQUIRE,
                                     __HIP_MEMORY_SCOPE_AGENT) < target)
                __builtin_amdgcn_s_sleep(1);
        }
        __syncthreads();
    }
}

// ---------------------------------------------------------------------------
// bf16 transpose via LDS: out[c,r] = in[r,c].  in: [R,C], out: [C,R].
// ---------------------------------------------------------------------------
__global__ __launch_bounds__(256) void transpose_bf16_kernel(
    const bf16* __restrict__ in, bf16* __restrict__ out, int R, int C) {
    __shared__ bf16 tile[32][33];
    int tx = threadIdx.x & 31;
    int ty = threadIdx.x >> 5;  // 0..7
    int r0 = blockIdx.y * 32;
    int c0 = blockIdx.x * 32;
#pragma unroll
    for (int i = 0; i < 32; i += 8)
        tile[ty + i][tx] = in[(size_t)(r0 + ty + i) * C + c0 + tx];
    __syncthreads();
#pragma unroll
    for (int i = 0; i < 32; i += 8)
        out[(size_t)(c0 + ty + i) * R + r0 + tx] = tile[tx][ty + i];
}

// ---------------------------------------------------------------------------
// Row softmax: attn_bf16[r] = softmax(E[r]).  One block per row.
// ---------------------------------------------------------------------------
__global__ __launch_bounds__(256) void softmax_row_kernel(
    const float* __restrict__ E, bf16* __restrict__ A, int cols) {
    __shared__ float red[256];
    const int tid = threadIdx.x;
    const float* e = E + (size_t)blockIdx.x * cols;
    bf16* a = A + (size_t)blockIdx.x * cols;

    float mx = -INFINITY;
    for (int j = tid; j < cols; j += 256) mx = fmaxf(mx, e[j]);
    red[tid] = mx;
    __syncthreads();
    for (int s = 128; s > 0; s >>= 1) {
        if (tid < s) red[tid] = fmaxf(red[tid], red[tid + s]);
        __syncthreads();
    }
    mx = red[0];
    __syncthreads();

    float sum = 0.0f;
    for (int j = tid; j < cols; j += 256) sum += __expf(e[j] - mx);
    red[tid] = sum;
    __syncthreads();
    for (int s = 128; s > 0; s >>= 1) {
        if (tid < s) red[tid] += red[tid + s];
        __syncthreads();
    }
    float inv = 1.0f / red[0];

    for (int j = tid; j < cols; j += 256) a[j] = (bf16)(__expf(e[j] - mx) * inv);
}

// ---------------------------------------------------------------------------
// out_bf16[m] = bf16(concat(hidden_state[m], context[m]))
// ---------------------------------------------------------------------------
__global__ __launch_bounds__(256) void concat_bf16_kernel(
    const float* __restrict__ hs, const float* __restrict__ ctx,
    bf16* __restrict__ out) {
    int m = blockIdx.x;
    bf16* o = out + (size_t)m * D2H;
    for (int j = threadIdx.x; j < D2H; j += 256)
        o[j] = (bf16)((j < HID) ? hs[(size_t)m * HID + j]
                                : ctx[(size_t)m * HID + j - HID]);
}

// ---------------------------------------------------------------------------
// In-place row log_softmax on [rows, cols] f32.  One block per row.
// ---------------------------------------------------------------------------
__global__ __launch_bounds__(256) void log_softmax_row_kernel(
    float* __restrict__ Y, int cols) {
    __shared__ float red[256];
    const int tid = threadIdx.x;
    float* y = Y + (size_t)blockIdx.x * cols;

    float mx = -INFINITY;
    for (int j = tid; j < cols; j += 256) mx = fmaxf(mx, y[j]);
    red[tid] = mx;
    __syncthreads();
    for (int s = 128; s > 0; s >>= 1) {
        if (tid < s) red[tid] = fmaxf(red[tid], red[tid + s]);
        __syncthreads();
    }
    mx = red[0];
    __syncthreads();

    float sum = 0.0f;
    for (int j = tid; j < cols; j += 256) sum += __expf(y[j] - mx);
    red[tid] = sum;
    __syncthreads();
    for (int s = 128; s > 0; s >>= 1) {
        if (tid < s) red[tid] += red[tid + s];
        __syncthreads();
    }
    float shift = mx + logf(red[0]);

    for (int j = tid; j < cols; j += 256) y[j] = y[j] - shift;
}

// ---------------------------------------------------------------------------
// Launch
// ---------------------------------------------------------------------------
extern "C" void kernel_launch(void* const* d_in, const int* in_sizes, int n_in,
                              void* d_out, int out_size, void* d_ws, size_t ws_size,
                              hipStream_t stream) {
    (void)in_sizes; (void)n_in; (void)out_size; (void)ws_size;

    const int*   loc     = (const int*)d_in[0];
    const int*   tim     = (const int*)d_in[1];
    // d_in[2] = target_len (compile-time constant TGT here)
    const float* emb_loc = (const float*)d_in[3];
    const float* emb_tim = (const float*)d_in[4];
    const float* Wih_enc = (const float*)d_in[5];
    const float* Whh_enc = (const float*)d_in[6];
    const float* bih_enc = (const float*)d_in[7];
    const float* bhh_enc = (const float*)d_in[8];
    const float* Wih_dec = (const float*)d_in[9];
    const float* Whh_dec = (const float*)d_in[10];
    const float* bih_dec = (const float*)d_in[11];
    const float* bhh_dec = (const float*)d_in[12];
    const float* W_fc    = (const float*)d_in[13];
    const float* b_fc    = (const float*)d_in[14];
    float*       out     = (float*)d_out;

    // ---- workspace carve (≈120 MB) ----
    char* p = (char*)d_ws;
    auto carve = [&](size_t bytes) {
        char* r = p;
        p += (bytes + 255) & ~(size_t)255;
        return (void*)r;
    };
    bf16*  x_bf     = (bf16*)carve((size_t)SEQ * INF_ * 2);
    bf16*  wihe_bf  = (bf16*)carve((size_t)H3 * INF_ * 2);
    bf16*  wihd_bf  = (bf16*)carve((size_t)H3 * INF_ * 2);
    float* gi_enc   = (float*)carve((size_t)SENC * H3 * 4);
    float* gi_dec   = (float*)carve((size_t)TGT * H3 * 4);
    float* hh       = (float*)carve((size_t)SENC * HID * 4);
    bf16*  hh_bf    = (bf16*)carve((size_t)SENC * HID * 2);
    float* hs       = (float*)carve((size_t)TGT * HID * 4);
    bf16*  hs_bf    = (bf16*)carve((size_t)TGT * HID * 2);
    bf16*  hhT      = (bf16*)carve((size_t)HID * SENC * 2);
    float* energies = (float*)carve((size_t)TGT * SENC * 4);
    bf16*  attn_bf  = (bf16*)carve((size_t)TGT * SENC * 2);
    float* ctx      = (float*)carve((size_t)TGT * HID * 4);
    bf16*  out_bf   = (bf16*)carve((size_t)TGT * D2H * 2);
    int*   sync_cnt = (int*)carve(2 * sizeof(int));
    float* hbuf     = (float*)carve(4 * HID * sizeof(float));  // enc A/B, dec A/B

    // 1) weight + embedding conversion to bf16
    {
        int n = H3 * INF_;
        cvt_f32_bf16_kernel<<<(n + 255) / 256, 256, 0, stream>>>(Wih_enc, wihe_bf, n);
        cvt_f32_bf16_kernel<<<(n + 255) / 256, 256, 0, stream>>>(Wih_dec, wihd_bf, n);
    }
    embed_kernel<<<SEQ, 256, 0, stream>>>(loc, tim, emb_loc, emb_tim, x_bf);

    // 2) input-gate GEMMs: gi = x @ Wih^T + bih   (WMMA bf16)
    gemm_abt_kernel<false, true><<<dim3(H3 / 128, SENC / 256), 256, 0, stream>>>(
        x_bf, wihe_bf, bih_enc, gi_enc, SENC, H3, INF_);
    gemm_abt_kernel<false, true><<<dim3(H3 / 128, TGT / 256), 256, 0, stream>>>(
        x_bf + (size_t)SENC * INF_, wihd_bf, bih_dec, gi_dec, TGT, H3, INF_);

    // 3) persistent parallel GRU scans (enc blocks 0..31, dec blocks 32..63)
    init_sync_kernel<<<1, 256, 0, stream>>>(sync_cnt, hbuf);
    gru_scan_kernel<<<2 * GRU_NB, GRU_THREADS, GRU_LDS_BYTES, stream>>>(
        gi_enc, Whh_enc, bhh_enc, gi_dec, Whh_dec, bhh_dec,
        hh, hh_bf, hs, hs_bf, sync_cnt, hbuf);

    // 4) attention: energies = hs @ hh^T  (WMMA)
    gemm_abt_kernel<false, false><<<dim3(SENC / 128, TGT / 256), 256, 0, stream>>>(
        hs_bf, hh_bf, nullptr, energies, TGT, SENC, HID);

    // 5) row softmax -> bf16 attention weights
    softmax_row_kernel<<<TGT, 256, 0, stream>>>(energies, attn_bf, SENC);

    // 6) context = attn @ hh = attn @ (hh^T)^T  (WMMA; needs hh transposed)
    transpose_bf16_kernel<<<dim3(HID / 32, SENC / 32), 256, 0, stream>>>(
        hh_bf, hhT, SENC, HID);
    gemm_abt_kernel<false, false><<<dim3(HID / 128, TGT / 256), 256, 0, stream>>>(
        attn_bf, hhT, nullptr, ctx, TGT, HID, SENC);

    // 7) concat + convert to bf16
    concat_bf16_kernel<<<TGT, 256, 0, stream>>>(hs, ctx, out_bf);

    // 8) FC GEMM: y = out @ W_fc^T + b_fc  (WMMA; W_fc converted f32->bf16
    //    in the fragment loader, streamed 4x from HBM)
    gemm_abt_kernel<true, true><<<dim3((NF + 127) / 128, TGT / 256), 256, 0, stream>>>(
        out_bf, (const void*)W_fc, b_fc, out, TGT, NF, D2H);

    // 9) in-place log_softmax on d_out rows
    log_softmax_row_kernel<<<TGT, 256, 0, stream>>>(out, NF);
}
